// GCN_22832046146035
// MI455X (gfx1250) — matrix operand order repaired and checked by reading, not compiled, not run on hardware
//
#include <hip/hip_runtime.h>
#include <hip/hip_bf16.h>

// ---------------- problem constants (match reference) ----------------
#define UU     100000
#define II     50000
#define NA     500
#define NS     50
#define DD     64
#define EEW    768
#define N_ER   200000
#define N_EAU  300000
#define N_EUAU 500000
#define N_EUAI 300000

typedef __attribute__((ext_vector_type(16))) __bf16        v16bf;
typedef __attribute__((ext_vector_type(8)))  float         v8f;
typedef __attribute__((ext_vector_type(8)))  unsigned int  v8u;

__device__ __forceinline__ float sigmoidf_(float x) {
  return 1.0f / (1.0f + __expf(-x));
}

// =====================================================================
// B-matrix fragment packing: B is [K x 64] f32 row-major.  For each
// K-step ks (32 rows) and N-tile tn (16 cols), emit the wave32 WMMA
// B-fragment (32 lanes x 16 bf16 = 32 lanes x 32B contiguous):
//   lanes 0-15  : col = tn*16+lane, K = ks*32 + 0..15
//   lanes 16-31 : col = tn*16+lane-16, K = ks*32 + 16..31
// Fragment f = ks*4+tn lives at Bp[f*32*8 + lane*8] (uints, bf16 pairs).
// =====================================================================
__global__ void pack_B_kernel(const float* __restrict__ B,
                              unsigned int* __restrict__ Bp, int K) {
  int t = blockIdx.x * blockDim.x + threadIdx.x;
  int nf = (K >> 5) * 4;
  if (t >= nf * 32) return;
  int f = t >> 5, lane = t & 31;
  int ks = f >> 2, tn = f & 3;
  int half = lane >> 4, lr = lane & 15;
  int col = tn * 16 + lr;
  int kb = ks * 32 + half * 16;
  unsigned int* dst = Bp + (size_t)f * 32 * 8 + (size_t)lane * 8;
#pragma unroll
  for (int p = 0; p < 8; ++p) {
    __bf16 lo = (__bf16)B[(size_t)(kb + 2 * p) * 64 + col];
    __bf16 hi = (__bf16)B[(size_t)(kb + 2 * p + 1) * 64 + col];
    unsigned short ulo = __builtin_bit_cast(unsigned short, lo);
    unsigned short uhi = __builtin_bit_cast(unsigned short, hi);
    dst[p] = (unsigned int)ulo | ((unsigned int)uhi << 16);
  }
}

// =====================================================================
// WMMA GEMM, N fixed at 64: C[M x 64] = A[M x K] @ B[K x 64].
// One wave32 computes a 16x64 strip: A fragment loaded ONCE per K-step
// (float4 loads + in-register cvt to bf16), 4 pre-packed B fragments,
// 4x v_wmma_f32_16x16x32_bf16 per step, f32 accumulate.
// A-fragment layout (16-bit A 16x32, contiguous k to match packed B):
//   lanes 0-15 : row=lane,    elems 0..15 = K+0..15
//   lanes 16-31: row=lane-16, elems 0..15 = K+16..31
// C/D layout: vreg i: lanes 0-15 -> M=i, lanes 16-31 -> M=i+8; N=lane%16.
// =====================================================================
__global__ void wmma_gemm_n64(const float* __restrict__ A,
                              const unsigned int* __restrict__ Bp,
                              float* __restrict__ C, int M, int K) {
  const int wave = (int)((blockIdx.x * (size_t)blockDim.x + threadIdx.x) >> 5);
  const int lane = threadIdx.x & 31;
  const int tilesM = (M + 15) >> 4;
  if (wave >= tilesM) return;
  const int half = lane >> 4;
  const int lr   = lane & 15;
  const int row  = wave * 16 + lr;
  const int rowc = row < M ? row : (M - 1);
  const float4* __restrict__ arow = (const float4*)(A + (size_t)rowc * K);
  const int ksteps = K >> 5;

  v8f acc0 = {}, acc1 = {}, acc2 = {}, acc3 = {};
  for (int ks = 0; ks < ksteps; ++ks) {
    const int fb = ks * 8 + half * 4;      // float4 index of K = ks*32 + half*16
    float4 a0 = arow[fb + 0];
    float4 a1 = arow[fb + 1];
    float4 a2 = arow[fb + 2];              // K + 8
    float4 a3 = arow[fb + 3];
    v16bf a;
    a[0]  = (__bf16)a0.x; a[1]  = (__bf16)a0.y; a[2]  = (__bf16)a0.z; a[3]  = (__bf16)a0.w;
    a[4]  = (__bf16)a1.x; a[5]  = (__bf16)a1.y; a[6]  = (__bf16)a1.z; a[7]  = (__bf16)a1.w;
    a[8]  = (__bf16)a2.x; a[9]  = (__bf16)a2.y; a[10] = (__bf16)a2.z; a[11] = (__bf16)a2.w;
    a[12] = (__bf16)a3.x; a[13] = (__bf16)a3.y; a[14] = (__bf16)a3.z; a[15] = (__bf16)a3.w;

    const v8u* __restrict__ bp = (const v8u*)Bp + (size_t)ks * 4 * 32 + lane;
    v16bf b0 = __builtin_bit_cast(v16bf, bp[0]);
    v16bf b1 = __builtin_bit_cast(v16bf, bp[32]);
    v16bf b2 = __builtin_bit_cast(v16bf, bp[64]);
    v16bf b3 = __builtin_bit_cast(v16bf, bp[96]);

    acc0 = __builtin_amdgcn_wmma_f32_16x16x32_bf16(false, a, false, b0, (short)0, acc0, false, false);
    acc1 = __builtin_amdgcn_wmma_f32_16x16x32_bf16(false, a, false, b1, (short)0, acc1, false, false);
    acc2 = __builtin_amdgcn_wmma_f32_16x16x32_bf16(false, a, false, b2, (short)0, acc2, false, false);
    acc3 = __builtin_amdgcn_wmma_f32_16x16x32_bf16(false, a, false, b3, (short)0, acc3, false, false);
  }

  float* __restrict__ crow = C + (size_t)(wave * 16 + half * 8) * 64 + lr;
  if (wave * 16 + 16 <= M) {
#pragma unroll
    for (int i = 0; i < 8; ++i) {
      crow[(size_t)i * 64 +  0] = acc0[i];
      crow[(size_t)i * 64 + 16] = acc1[i];
      crow[(size_t)i * 64 + 32] = acc2[i];
      crow[(size_t)i * 64 + 48] = acc3[i];
    }
  } else {
#pragma unroll
    for (int i = 0; i < 8; ++i) {
      int r = wave * 16 + half * 8 + i;
      if (r < M) {
        crow[(size_t)i * 64 +  0] = acc0[i];
        crow[(size_t)i * 64 + 16] = acc1[i];
        crow[(size_t)i * 64 + 32] = acc2[i];
        crow[(size_t)i * 64 + 48] = acc3[i];
      }
    }
  }
}

// =====================================================================
// Edge kernels: 64 threads per edge (one per feature column), atomicAdd
// segment-sum into a [nDst x 192] accumulator (3 concatenated 64-blocks).
// =====================================================================
__global__ void edge_aspect_kernel(const int* __restrict__ src,
                                   const int* __restrict__ dst,
                                   const float* __restrict__ a_fe,
                                   const float* __restrict__ SF,
                                   const float* __restrict__ c_asp,
                                   float* __restrict__ outbuf,
                                   int nE, int blockOff) {
  long long t = (long long)blockIdx.x * blockDim.x + threadIdx.x;
  int e = (int)(t >> 6), j = (int)(t & 63);
  if (e >= nE) return;
  int s = src[e], d = dst[e];
  float v = (a_fe[(size_t)s * DD + j] + SF[(size_t)e * DD + j]) * c_asp[s];
  atomicAdd(&outbuf[(size_t)d * 192 + blockOff + j], v);
}

__global__ void edge_gated_kernel(const int* __restrict__ src,
                                  const int* __restrict__ dst,
                                  const int* __restrict__ asp3,  // [nE x 3]
                                  const float* __restrict__ fe,
                                  const float* __restrict__ a_fe1,
                                  const float* __restrict__ sgate, // [NS x 64]
                                  const float* __restrict__ c_src,
                                  float* __restrict__ outbuf, int nE) {
  long long t = (long long)blockIdx.x * blockDim.x + threadIdx.x;
  int e = (int)(t >> 6), j = (int)(t & 63);
  if (e >= nE) return;
  int s  = src[e], d = dst[e];
  int a0 = asp3[3 * (size_t)e + 0];
  int a1 = asp3[3 * (size_t)e + 1];
  int a2 = asp3[3 * (size_t)e + 2];
  float g = sigmoidf_(sgate[(size_t)a1 * DD + j] + sgate[(size_t)a2 * DD + j]);
  float v = (fe[(size_t)s * DD + j] + a_fe1[(size_t)a0 * DD + j]) * g * c_src[s];
  atomicAdd(&outbuf[(size_t)d * 192 + 64 + j], v);
}

__global__ void edge_plain_kernel(const int* __restrict__ src,
                                  const int* __restrict__ dst,
                                  const int* __restrict__ asp,
                                  const float* __restrict__ fee,
                                  const float* __restrict__ a_fe1,
                                  const float* __restrict__ c_src,
                                  float* __restrict__ outbuf, int nE) {
  long long t = (long long)blockIdx.x * blockDim.x + threadIdx.x;
  int e = (int)(t >> 6), j = (int)(t & 63);
  if (e >= nE) return;
  int s = src[e], d = dst[e], a = asp[e];
  float v = (fee[(size_t)s * DD + j] + a_fe1[(size_t)a * DD + j]) * c_src[s];
  atomicAdd(&outbuf[(size_t)d * 192 + 128 + j], v);
}

__global__ void scale3_kernel(float* __restrict__ buf,
                              const float* __restrict__ c0,
                              const float* __restrict__ c1,
                              const float* __restrict__ c2, int nRows) {
  long long t = (long long)blockIdx.x * blockDim.x + threadIdx.x;
  int r = (int)(t / 192), j = (int)(t % 192);
  if (r >= nRows) return;
  const float* c = (j < 64) ? c0 : ((j < 128) ? c1 : c2);
  buf[(size_t)r * 192 + j] *= c[r];
}

// one 256-thread block per review edge
__global__ void review_scatter_kernel(const int* __restrict__ src,
                                      const int* __restrict__ dst,
                                      const int* __restrict__ score,
                                      const float* __restrict__ from_a,
                                      const float* __restrict__ rproj,
                                      const float* __restrict__ semb, // [5 x 256]
                                      const float* __restrict__ c_src,
                                      float* __restrict__ out, size_t rowOff) {
  int e = blockIdx.x;
  int j = threadIdx.x;
  int s = src[e], d = dst[e], sc = score[e];
  float base = (j < 192) ? from_a[(size_t)s * 192 + j]
                         : rproj[(size_t)e * DD + (j - 192)];
  float v = base * sigmoidf_(semb[(size_t)sc * 256 + j]) * c_src[s];
  atomicAdd(&out[rowOff + (size_t)d * 256 + j], v);
}

__global__ void final_scale_kernel(float* __restrict__ out,
                                   const float* __restrict__ c,
                                   int nRows, size_t rowOff) {
  long long t = (long long)blockIdx.x * blockDim.x + threadIdx.x;
  int r = (int)(t >> 8), j = (int)(t & 255);
  if (r >= nRows) return;
  out[rowOff + (size_t)r * 256 + j] *= c[r];
}

// =====================================================================
static inline int cdiv_i(long long a, long long b) { return (int)((a + b - 1) / b); }

static void launch_pack(const float* B, unsigned int* Bp, int K, hipStream_t s) {
  int threads = (K >> 5) * 4 * 32;
  hipLaunchKernelGGL(pack_B_kernel, dim3(cdiv_i(threads, 256)), dim3(256), 0, s, B, Bp, K);
}
static void launch_gemm(const float* A, const unsigned int* Bp, float* C,
                        int M, int K, hipStream_t stream) {
  long long waves = (M + 15) / 16;   // one wave per 16x64 strip
  hipLaunchKernelGGL(wmma_gemm_n64, dim3(cdiv_i(waves * 32, 256)), dim3(256),
                     0, stream, A, Bp, C, M, K);
}

extern "C" void kernel_launch(void* const* d_in, const int* in_sizes, int n_in,
                              void* d_out, int out_size, void* d_ws, size_t ws_size,
                              hipStream_t stream) {
  (void)in_sizes; (void)n_in; (void)ws_size;
  const float* feature       = (const float*)d_in[0];
  const float* weight        = (const float*)d_in[1];
  const float* aspect_feat   = (const float*)d_in[2];
  const float* sentiment     = (const float*)d_in[3];
  const float* W_review      = (const float*)d_in[4];
  const float* W_review_r    = (const float*)d_in[5];
  const float* W_aspect      = (const float*)d_in[6];
  const float* W_aspect_r    = (const float*)d_in[7];
  const float* W_sent        = (const float*)d_in[8];
  const float* W_sent_r      = (const float*)d_in[9];
  const float* W_s1          = (const float*)d_in[10];
  const float* W_s2          = (const float*)d_in[11];
  const float* score_emb     = (const float*)d_in[12];
  const float* score_r_emb   = (const float*)d_in[13];
  const float* review_feat   = (const float*)d_in[14];
  const float* review_r_feat = (const float*)d_in[15];
  const float* au_sent_feat  = (const float*)d_in[16];
  const float* ai_sent_feat  = (const float*)d_in[17];
  const float* cau_aspect    = (const float*)d_in[18];
  const float* cai_aspect    = (const float*)d_in[19];
  const float* cau_user      = (const float*)d_in[20];
  const float* c_uau_user    = (const float*)d_in[21];
  const float* c_uau_r_user  = (const float*)d_in[22];
  const float* c_uai_user    = (const float*)d_in[23];
  const float* cur_user      = (const float*)d_in[24];
  const float* cir_user      = (const float*)d_in[25];
  const float* cai_item      = (const float*)d_in[26];
  const float* c_iai_item    = (const float*)d_in[27];
  const float* c_iai_r_item  = (const float*)d_in[28];
  const float* c_iau_item    = (const float*)d_in[29];
  const float* cir_item      = (const float*)d_in[30];
  const float* cur_item      = (const float*)d_in[31];
  const int* review_src   = (const int*)d_in[32];
  const int* review_dst   = (const int*)d_in[33];
  const int* review_score = (const int*)d_in[34];
  const int* review_r_src = (const int*)d_in[35];
  const int* review_r_dst = (const int*)d_in[36];
  const int* review_r_scr = (const int*)d_in[37];
  const int* au_src  = (const int*)d_in[38];
  const int* au_dst  = (const int*)d_in[39];
  const int* ai_src  = (const int*)d_in[40];
  const int* ai_dst  = (const int*)d_in[41];
  const int* uau_src = (const int*)d_in[42];
  const int* uau_dst = (const int*)d_in[43];
  const int* uau_asp = (const int*)d_in[44];
  const int* iai_src = (const int*)d_in[45];
  const int* iai_dst = (const int*)d_in[46];
  const int* iai_asp = (const int*)d_in[47];
  const int* uai_src = (const int*)d_in[48];
  const int* uai_dst = (const int*)d_in[49];
  const int* uai_asp = (const int*)d_in[50];
  const int* iau_src = (const int*)d_in[51];
  const int* iau_dst = (const int*)d_in[52];
  const int* iau_asp = (const int*)d_in[53];

  float* out = (float*)d_out;
  const float* fe_u  = feature;
  const float* fe_i  = feature + (size_t)UU * DD;
  const float* fee_u = weight;
  const float* fee_i = weight + (size_t)UU * DD;

  // ---- workspace layout (floats). SF buffers are reused for r_rev / r_rr
  //      after the aspect edge kernels have consumed them (stream-ordered).
  float* ws = (float*)d_ws;
  size_t o = 0;
  float* a_fe      = ws + o; o += (size_t)NA * DD;
  float* a_fe1     = ws + o; o += (size_t)NA * DD;
  float* s1        = ws + o; o += (size_t)NS * DD;
  float* s2        = ws + o; o += (size_t)NS * DD;
  float* from_a_u  = ws + o; o += (size_t)UU * 192;
  float* from_a_i  = ws + o; o += (size_t)II * 192;
  float* bufA      = ws + o; o += (size_t)N_EAU * DD; // SF_au -> r_rev
  float* bufB      = ws + o; o += (size_t)N_EAU * DD; // SF_ai -> r_rr
  // packed bf16 B-fragment buffers (uints)
  unsigned int* wsu = (unsigned int*)(ws + o);
  size_t uo = 0;
  const size_t PB64  = 2 * 4 * 32 * 8;   // K=64  packed size
  const size_t PB768 = 24 * 4 * 32 * 8;  // K=768 packed size
  unsigned int* Bp_aspect   = wsu + uo; uo += PB64;
  unsigned int* Bp_aspect_r = wsu + uo; uo += PB64;
  unsigned int* Bp_s1       = wsu + uo; uo += PB64;
  unsigned int* Bp_s2       = wsu + uo; uo += PB64;
  unsigned int* Bp_sent     = wsu + uo; uo += PB64;
  unsigned int* Bp_sent_r   = wsu + uo; uo += PB64;
  unsigned int* Bp_review   = wsu + uo; uo += PB768;
  unsigned int* Bp_review_r = wsu + uo; uo += PB768;

  // ---- zero accumulators (every call: determinism)
  hipMemsetAsync(from_a_u, 0, (size_t)UU * 192 * sizeof(float), stream);
  hipMemsetAsync(from_a_i, 0, (size_t)II * 192 * sizeof(float), stream);
  hipMemsetAsync(out, 0, (size_t)out_size * sizeof(float), stream);

  // ---- pack weight matrices into WMMA B-fragment layout (bf16)
  launch_pack(W_aspect,   Bp_aspect,   DD,  stream);
  launch_pack(W_aspect_r, Bp_aspect_r, DD,  stream);
  launch_pack(W_s1,       Bp_s1,       DD,  stream);
  launch_pack(W_s2,       Bp_s2,       DD,  stream);
  launch_pack(W_sent,     Bp_sent,     DD,  stream);
  launch_pack(W_sent_r,   Bp_sent_r,   DD,  stream);
  launch_pack(W_review,   Bp_review,   EEW, stream);
  launch_pack(W_review_r, Bp_review_r, EEW, stream);

  // ---- small projections (WMMA)
  launch_gemm(aspect_feat, Bp_aspect,   a_fe,  NA, DD, stream);
  launch_gemm(aspect_feat, Bp_aspect_r, a_fe1, NA, DD, stream);
  launch_gemm(sentiment,   Bp_s1,       s1,    NS, DD, stream);
  launch_gemm(sentiment,   Bp_s2,       s2,    NS, DD, stream);

  // ---- sentiment-edge projections (WMMA, 300k x 64 @ 64 x 64)
  launch_gemm(au_sent_feat, Bp_sent,   bufA, N_EAU, DD, stream);
  launch_gemm(ai_sent_feat, Bp_sent_r, bufB, N_EAU, DD, stream);

  // ---- aspect -> user/item aggregation (h_u, h_i)
  {
    int blocks = cdiv_i((long long)N_EAU * 64, 256);
    hipLaunchKernelGGL(edge_aspect_kernel, dim3(blocks), dim3(256), 0, stream,
                       au_src, au_dst, a_fe, bufA, cau_aspect, from_a_u, N_EAU, 0);
    hipLaunchKernelGGL(edge_aspect_kernel, dim3(blocks), dim3(256), 0, stream,
                       ai_src, ai_dst, a_fe, bufB, cai_aspect, from_a_i, N_EAU, 0);
  }
  // ---- gated two-hop (h1_u, h2_i)
  {
    int blocks = cdiv_i((long long)N_EUAU * 64, 256);
    hipLaunchKernelGGL(edge_gated_kernel, dim3(blocks), dim3(256), 0, stream,
                       uau_src, uau_dst, uau_asp, fe_u, a_fe1, s1, c_uau_user,
                       from_a_u, N_EUAU);
    hipLaunchKernelGGL(edge_gated_kernel, dim3(blocks), dim3(256), 0, stream,
                       iai_src, iai_dst, iai_asp, fe_i, a_fe1, s2, c_iai_item,
                       from_a_i, N_EUAU);
  }
  // ---- cross-type (h3_i, h3_u)
  {
    int blocks = cdiv_i((long long)N_EUAI * 64, 256);
    hipLaunchKernelGGL(edge_plain_kernel, dim3(blocks), dim3(256), 0, stream,
                       uai_src, uai_dst, uai_asp, fee_u, a_fe1, c_uai_user,
                       from_a_i, N_EUAI);
    hipLaunchKernelGGL(edge_plain_kernel, dim3(blocks), dim3(256), 0, stream,
                       iau_src, iau_dst, iau_asp, fee_i, a_fe1, c_iau_item,
                       from_a_u, N_EUAI);
  }
  // ---- per-destination scaling of concatenated blocks
  hipLaunchKernelGGL(scale3_kernel, dim3(cdiv_i((long long)UU * 192, 256)),
                     dim3(256), 0, stream,
                     from_a_u, cau_user, c_uau_r_user, c_uai_user, UU);
  hipLaunchKernelGGL(scale3_kernel, dim3(cdiv_i((long long)II * 192, 256)),
                     dim3(256), 0, stream,
                     from_a_i, cai_item, c_iai_r_item, c_iau_item, II);

  // ---- big review GEMMs (WMMA, 200k x 768 @ 768 x 64), reuse SF buffers
  launch_gemm(review_feat,   Bp_review,   bufA, N_ER, EEW, stream);
  launch_gemm(review_r_feat, Bp_review_r, bufB, N_ER, EEW, stream);

  // ---- final review scatter into output
  hipLaunchKernelGGL(review_scatter_kernel, dim3(N_ER), dim3(256), 0, stream,
                     review_src, review_dst, review_score,
                     from_a_i, bufA, score_emb, cur_item, out, (size_t)0);
  hipLaunchKernelGGL(review_scatter_kernel, dim3(N_ER), dim3(256), 0, stream,
                     review_r_src, review_r_dst, review_r_scr,
                     from_a_u, bufB, score_r_emb, cir_user, out,
                     (size_t)UU * 256);

  hipLaunchKernelGGL(final_scale_kernel, dim3(cdiv_i((long long)UU * 256, 256)),
                     dim3(256), 0, stream, out, cur_user, UU, (size_t)0);
  hipLaunchKernelGGL(final_scale_kernel, dim3(cdiv_i((long long)II * 256, 256)),
                     dim3(256), 0, stream, out, cir_item, II, (size_t)UU * 256);
}
